// CQT_64733747085470
// MI455X (gfx1250) — matrix operand order repaired and checked by reading, not compiled, not run on hardware
//
#include <hip/hip_runtime.h>
#include <hip/hip_bf16.h>
#include <math.h>

// ---------------- problem constants (from the reference) ----------------
#define HOP      512
#define FFTLEN   2048
#define FBINS    1025          // fftLen/2 + 1
#define NBINS    84
#define MPAD     96            // 84 padded to 6 WMMA row-tiles
#define NFRAMES  16381         // (2^23 - 2048)/512 + 1
#define NPAIRS   512           // 1024 16-frame tiles, processed in pairs
#define KTILES   (FFTLEN / 32) // 64 WMMA K-steps

typedef __attribute__((ext_vector_type(16))) _Float16 v16h;
typedef __attribute__((ext_vector_type(8)))  float    v8f;

// ---------------------------------------------------------------------------
// Kernel 1: fold CQT projection into Fourier kernels (matrix associativity):
//   Ar[m][k] = sum_p kr[m][p]*wcos[p][k] - ki[m][p]*wsin[p][k]
//   Ai[m][k] = sum_p kr[m][p]*wsin[p][k] + ki[m][p]*wcos[p][k]
// Stored f16 in a 16x32 A-fragment lane order of this code's own (contiguous K, B matches), with the
// re/im fragments interleaved per K-step so the GEMM reads both off ONE base
// register with immediate offsets:
//   Aq[(m_tile*64 + k_tile)*1024 + sel*512 + lane*16 + j], sel: 0=re, 1=im
//   element j of lane L (h=L>>4, mr=L&15): K = h*16 + j
// ---------------------------------------------------------------------------
__global__ __launch_bounds__(256) void cqt_make_A(const float* __restrict__ kr,
                                                  const float* __restrict__ ki,
                                                  const float* __restrict__ wcos,
                                                  const float* __restrict__ wsin,
                                                  _Float16* __restrict__ Aq) {
    const int gid = blockIdx.x * blockDim.x + threadIdx.x;  // 96*2048 threads
    const int m = gid >> 11;        // 0..95  (uniform within a 256-block)
    const int k = gid & (FFTLEN - 1);

    float ar = 0.0f, ai = 0.0f;
    if (m < NBINS) {
        const float* krp = kr + (size_t)m * FBINS;
        const float* kip = ki + (size_t)m * FBINS;
        for (int p = 0; p < FBINS; ++p) {
            const float c = wcos[(size_t)p * FFTLEN + k];
            const float s = wsin[(size_t)p * FFTLEN + k];
            const float r = krp[p];
            const float im = kip[p];
            ar = fmaf(r, c, ar);  ar = fmaf(-im, s, ar);
            ai = fmaf(r, s, ai);  ai = fmaf(im, c, ai);
        }
    }

    // swizzled store position
    const int m_tile = m >> 4, mr = m & 15;
    const int k_tile = k >> 5, kk = k & 31;
    const int g = kk >> 3;                    // 8-element group within 32
    const int h = kk >> 4;                    // lane half
    const int j = kk & 15;                    // element within v16h
    const int lane = h * 16 + mr;
    const size_t base = (size_t)(m_tile * KTILES + k_tile) * 1024 + lane * 16 + j;
    Aq[base]       = (_Float16)ar;   // re fragment
    Aq[base + 512] = (_Float16)ai;   // im fragment
}

// ---------------------------------------------------------------------------
// Kernel 2: convert x (f32) -> f16 once, so the GEMM streams half the bytes.
// ---------------------------------------------------------------------------
__global__ __launch_bounds__(256) void cqt_cvt_x(const float* __restrict__ x,
                                                 _Float16* __restrict__ x16) {
    const int i = blockIdx.x * blockDim.x + threadIdx.x;
    x16[i] = (_Float16)x[i];
}

// fallback helper: load 16 consecutive f32 and convert to a v16h B fragment
__device__ __forceinline__ v16h load_cvt16(const float* bp) {
    const float4 f0 = ((const float4*)bp)[0];
    const float4 f1 = ((const float4*)bp)[1];
    const float4 f2 = ((const float4*)bp)[2];
    const float4 f3 = ((const float4*)bp)[3];
    v16h b;
    b[0]  = (_Float16)f0.x; b[1]  = (_Float16)f0.y;
    b[2]  = (_Float16)f0.z; b[3]  = (_Float16)f0.w;
    b[4]  = (_Float16)f1.x; b[5]  = (_Float16)f1.y;
    b[6]  = (_Float16)f1.z; b[7]  = (_Float16)f1.w;
    b[8]  = (_Float16)f2.x; b[9]  = (_Float16)f2.y;
    b[10] = (_Float16)f2.z; b[11] = (_Float16)f2.w;
    b[12] = (_Float16)f3.x; b[13] = (_Float16)f3.y;
    b[14] = (_Float16)f3.z; b[15] = (_Float16)f3.w;
    return b;
}

// ---------------------------------------------------------------------------
// Kernel 3: fused GEMM + magnitude.
//   D[96, 16381] (re & im) = A[96, 2048] @ framesT[2048, 16381]
//   framesT[k][f] = x[f*HOP + k]   (frames implicit, never materialized)
// Block = 192 threads = 6 wave32s; wave w owns row-tile m_tile = w,
// block b owns frame-tile pair {2b, 2b+1}. 64 K-steps, 4 WMMAs per step
// (re/im x 2 N-tiles) against 8 b128 loads -> 2:1 vmem:wmma issue ratio,
// 4 independent accumulate chains for WMMA ILP.
// ---------------------------------------------------------------------------
template <bool PRE>
__global__ __launch_bounds__(192) void cqt_gemm(const _Float16* __restrict__ Aq,
                                                const _Float16* __restrict__ x16,
                                                const float* __restrict__ x32,
                                                float* __restrict__ out) {
    const int lane   = threadIdx.x & 31;
    const int m_tile = threadIdx.x >> 5;   // 0..5
    const int h  = lane >> 4;              // lane half
    const int mr = lane & 15;

    // B columns (frames) for this lane. f0 is always in range; only the
    // second tile of the last pair can run past the end.
    const int f0 = blockIdx.x * 32 + mr;         // <= 16367 < NFRAMES
    int f1 = f0 + 16;
    if (f1 > NFRAMES - 1) f1 = NFRAMES - 1;
    const size_t koff = (size_t)(h * 16);        // K-half base within 32

    // interleaved A fragment stream (re @ +0, im @ +512, step 1024 f16/iter)
    const _Float16* aq = Aq + (size_t)m_tile * KTILES * 1024 + lane * 16;

    v8f accr0 = {}, acci0 = {};
    v8f accr1 = {}, acci1 = {};

    if constexpr (PRE) {
        const _Float16* b0p = x16 + (size_t)f0 * HOP + koff;
        const _Float16* b1p = x16 + (size_t)f1 * HOP + koff;
#pragma unroll 2
        for (int kt = 0; kt < KTILES; ++kt) {
            const v16h a_r = *(const v16h*)(aq);
            const v16h a_i = *(const v16h*)(aq + 512);
            aq += 1024;
            const v16h b0 = *(const v16h*)b0p;  b0p += 32;
            const v16h b1 = *(const v16h*)b1p;  b1p += 32;
            accr0 = __builtin_amdgcn_wmma_f32_16x16x32_f16(
                false, a_r, false, b0, (short)0, accr0, false, false);
            acci0 = __builtin_amdgcn_wmma_f32_16x16x32_f16(
                false, a_i, false, b0, (short)0, acci0, false, false);
            accr1 = __builtin_amdgcn_wmma_f32_16x16x32_f16(
                false, a_r, false, b1, (short)0, accr1, false, false);
            acci1 = __builtin_amdgcn_wmma_f32_16x16x32_f16(
                false, a_i, false, b1, (short)0, acci1, false, false);
        }
    } else {
        const float* b0p = x32 + (size_t)f0 * HOP + koff;
        const float* b1p = x32 + (size_t)f1 * HOP + koff;
        for (int kt = 0; kt < KTILES; ++kt) {
            const v16h a_r = *(const v16h*)(aq);
            const v16h a_i = *(const v16h*)(aq + 512);
            aq += 1024;
            const v16h b0 = load_cvt16(b0p);  b0p += 32;
            const v16h b1 = load_cvt16(b1p);  b1p += 32;
            accr0 = __builtin_amdgcn_wmma_f32_16x16x32_f16(
                false, a_r, false, b0, (short)0, accr0, false, false);
            acci0 = __builtin_amdgcn_wmma_f32_16x16x32_f16(
                false, a_i, false, b0, (short)0, acci0, false, false);
            accr1 = __builtin_amdgcn_wmma_f32_16x16x32_f16(
                false, a_r, false, b1, (short)0, accr1, false, false);
            acci1 = __builtin_amdgcn_wmma_f32_16x16x32_f16(
                false, a_i, false, b1, (short)0, acci1, false, false);
        }
    }

    // Epilogue: magnitude + guarded store.
    // C/D layout: VGPR r -> M = h*8 + r, N = lane&15.
    const int fcol1 = f0 + 16;   // un-clamped column for tile 1
#pragma unroll
    for (int r = 0; r < 8; ++r) {
        const int bin = m_tile * 16 + h * 8 + r;
        if (bin < NBINS) {
            const float re0 = accr0[r];
            const float im0 = acci0[r];
            out[(size_t)bin * NFRAMES + f0] = sqrtf(re0 * re0 + im0 * im0);
            if (fcol1 < NFRAMES) {
                const float re1 = accr1[r];
                const float im1 = acci1[r];
                out[(size_t)bin * NFRAMES + fcol1] = sqrtf(re1 * re1 + im1 * im1);
            }
        }
    }
}

// ---------------------------------------------------------------------------
// Host launcher. Inputs (setup_inputs order): x, wcos, wsin, kr, ki.
// Output: [1, 84, 16381] f32.
// ---------------------------------------------------------------------------
extern "C" void kernel_launch(void* const* d_in, const int* in_sizes, int n_in,
                              void* d_out, int out_size, void* d_ws, size_t ws_size,
                              hipStream_t stream) {
    const float* x    = (const float*)d_in[0];
    const float* wcos = (const float*)d_in[1];
    const float* wsin = (const float*)d_in[2];
    const float* kr   = (const float*)d_in[3];
    const float* ki   = (const float*)d_in[4];
    float* out = (float*)d_out;

    const size_t nA = (size_t)2 * MPAD * FFTLEN;   // interleaved re/im f16
    const size_t nX = (size_t)8388608;             // 2^23 samples

    _Float16* Aq  = (_Float16*)d_ws;
    _Float16* x16 = Aq + nA;

    const size_t need_pre = (nA + nX) * sizeof(_Float16);
    const bool pre = (ws_size >= need_pre);

    // Fold kr/ki into the Fourier kernels (96x2048 combined complex kernel).
    cqt_make_A<<<(MPAD * FFTLEN) / 256, 256, 0, stream>>>(kr, ki, wcos, wsin, Aq);

    if (pre) {
        cqt_cvt_x<<<(int)(nX / 256), 256, 0, stream>>>(x, x16);
        cqt_gemm<true><<<NPAIRS, 192, 0, stream>>>(Aq, x16, x, out);
    } else {
        cqt_gemm<false><<<NPAIRS, 192, 0, stream>>>(Aq, nullptr, x, out);
    }
}
